// TFN_2259152797791
// MI455X (gfx1250) — matrix operand order repaired and checked by reading, not compiled, not run on hardware
//
#include <hip/hip_runtime.h>
#include <math.h>

// ---------------------------------------------------------------------------
// TFN (SE(3)-equivariant GCN) forward for MI455X / gfx1250.
//
// Dominant compute = radial-MLP w3 GEMM ([E,32] x [32, mo*mi*nf], ~83% of
// ~120 GFLOP) -> mapped to v_wmma_f32_16x16x32_bf16 (bf16 in, f32 accum).
// Per-edge basis/message contractions fused in the same 1-wave/16-edge-tile
// kernel via LDS staging; segment-mean via global f32 atomics; small
// node-wise kernels for gnorm/self/FC.  The conv kernel is templated on the
// 24 static (mo,mi,P,Q,nf) shapes so all inner loops fully unroll.
//
// Input leaf indices (dict insertion order at top level, sorted-key pytree
// flatten for nested dicts):
//   0:f 1:d 2:w
//   3..18   : basis["di,do"]  index = 3 + di*4 + do   [E,2do+1,2di+1,nf]
//   19..43  : convs[0] radial keys "0,0","1,0","2,0","3,0" (entry=do),
//             leaves sorted {b1,b2,b3,w1,w2,w3}; self[0] at 43
//   44..143 : convs[1] radial (do*4+di)*6 + 44; self d at 140+d
//   144..243: convs[2] (same, +100)
//   244..268: convs[3] radial entry=di (do=0); self[0] at 268
//   269..272: fc {b1,b2,w1,w2}
//   273..296: norms[i][d] {b,w}: 273 + i*8 + d*2 (+0=b,+1=w)
//   297:src 298:dst
// ---------------------------------------------------------------------------

#define NN 4096
#define NE 65536

typedef __attribute__((ext_vector_type(16))) __bf16 v16bf;
typedef __attribute__((ext_vector_type(8)))  float  v8f;

static __device__ __forceinline__ __bf16 tobf(float x) { return (__bf16)x; }

// ---------------- degree counts ----------------
__global__ void k_deg(const int* __restrict__ dst, float* __restrict__ cnt) {
  int e = blockIdx.x * 256 + threadIdx.x;
  if (e < NE) atomicAdd(&cnt[dst[e]], 1.0f);
}

// ---------------- edge features, packed into WMMA A-operand order ----------
// A (16x32 bf16): lane l<16: M=l, K halves j: j<8 -> K=j else K=j+8
//                 lane l>=16: M=l-16,            j<8 -> K=j+8 else K=j+16
__global__ void k_featpack(const float* __restrict__ d, const float* __restrict__ w,
                           __bf16* __restrict__ featA) {
  int tid = blockIdx.x * 128 + threadIdx.x;      // tile*32 + lane
  if (tid >= NE * 2) return;
  int t = tid >> 5, l = tid & 31;
  int e = t * 16 + (l & 15);
  float dx = d[e*3], dy = d[e*3+1], dz = d[e*3+2];
  float fv[5] = { w[e*4], w[e*4+1], w[e*4+2], w[e*4+3],
                  sqrtf(dx*dx + dy*dy + dz*dz) };
  __bf16* o = featA + (size_t)tid * 16;
#pragma unroll
  for (int j = 0; j < 16; j++) {
    int k = (l < 16) ? (j < 8 ? j : j + 8) : (j < 8 ? j + 8 : j + 16);
    o[j] = tobf(k < 5 ? fv[k] : 0.0f);
  }
}

// ---------------- pack w1(pad K to 32), w2, w3 into WMMA B-operand order ---
// B (32x16 bf16) tile: lane l: N = l&15; l<16: j<8 -> K=j else K=j+8; l>=16: j<8 -> K=j+8 else K=j+16
__global__ void k_wpack(const float* __restrict__ w1, const float* __restrict__ w2,
                        const float* __restrict__ w3, int ncol3,
                        __bf16* __restrict__ w1p, __bf16* __restrict__ w2p,
                        __bf16* __restrict__ w3p) {
  int tid = blockIdx.x * 128 + threadIdx.x;
  int nt3 = ncol3 >> 4;
  if (tid >= (4 + nt3) * 32) return;
  int t = tid >> 5, l = tid & 31;
  int col = l & 15;
#pragma unroll
  for (int j = 0; j < 16; j++) {
    int k = (l < 16) ? (j < 8 ? j : j + 8) : (j < 8 ? j + 8 : j + 16);
    if (t < 2) {
      int c = t*16 + col;
      w1p[(t*32 + l)*16 + j] = tobf(k < 5 ? w1[k*32 + c] : 0.0f);
    } else if (t < 4) {
      int tt = t - 2, c = tt*16 + col;
      w2p[(tt*32 + l)*16 + j] = tobf(w2[k*32 + c]);
    } else {
      int tt = t - 4, c = tt*16 + col;
      w3p[((size_t)tt*32 + l)*16 + j] = tobf(w3[(size_t)k*ncol3 + c]);
    }
  }
}

// ---------------- radial MLP hidden: relu(LN(relu(LN(feat@w1+b1))@w2+b2)) --
// 1 wave / 16-edge tile; emits h2 pre-packed as the next WMMA A operand.
__global__ __launch_bounds__(32)
void k_radial_hidden(const __bf16* __restrict__ featA,
                     const __bf16* __restrict__ w1p, const __bf16* __restrict__ w2p,
                     const float* __restrict__ b1,  const float* __restrict__ b2,
                     __bf16* __restrict__ h2A) {
  __shared__ float hid[16 * 32];
  __shared__ float mu[16], rs[16];
  int lane = threadIdx.x, tile = blockIdx.x;
  int col = lane & 15, rh = (lane >> 4) * 8, myrow = lane & 15;
  v16bf a = *(const v16bf*)(featA + (size_t)(tile*32 + lane)*16);

#pragma unroll
  for (int t = 0; t < 2; t++) {
    v16bf b = *(const v16bf*)(w1p + (t*32 + lane)*16);
    v8f c = {};
    c = __builtin_amdgcn_wmma_f32_16x16x32_bf16(false, a, false, b, (short)0, c, false, false);
    float bb = b1[t*16 + col];
#pragma unroll
    for (int v = 0; v < 8; v++) hid[(rh + v)*32 + t*16 + col] = c[v] + bb;
  }
  __syncthreads();
  if (lane < 16) {
    float s = 0.f;  for (int k = 0; k < 32; k++) s += hid[lane*32 + k];
    float m = s / 32.f, v2 = 0.f;
    for (int k = 0; k < 32; k++) { float dv = hid[lane*32+k] - m; v2 += dv*dv; }
    mu[lane] = m; rs[lane] = rsqrtf(v2 / 32.f + 1e-5f);
  }
  __syncthreads();
  v16bf a2;
#pragma unroll
  for (int j = 0; j < 16; j++) {
    int k = (lane < 16) ? (j < 8 ? j : j + 8) : (j < 8 ? j + 8 : j + 16);
    float x = (hid[myrow*32 + k] - mu[myrow]) * rs[myrow];
    a2[j] = tobf(fmaxf(x, 0.f));
  }
  __syncthreads();
#pragma unroll
  for (int t = 0; t < 2; t++) {
    v16bf b = *(const v16bf*)(w2p + (t*32 + lane)*16);
    v8f c = {};
    c = __builtin_amdgcn_wmma_f32_16x16x32_bf16(false, a2, false, b, (short)0, c, false, false);
    float bb = b2[t*16 + col];
#pragma unroll
    for (int v = 0; v < 8; v++) hid[(rh + v)*32 + t*16 + col] = c[v] + bb;
  }
  __syncthreads();
  if (lane < 16) {
    float s = 0.f;  for (int k = 0; k < 32; k++) s += hid[lane*32 + k];
    float m = s / 32.f, v2 = 0.f;
    for (int k = 0; k < 32; k++) { float dv = hid[lane*32+k] - m; v2 += dv*dv; }
    mu[lane] = m; rs[lane] = rsqrtf(v2 / 32.f + 1e-5f);
  }
  __syncthreads();
  __bf16* o = h2A + (size_t)(tile*32 + lane)*16;
#pragma unroll
  for (int j = 0; j < 16; j++) {
    int k = (lane < 16) ? (j < 8 ? j : j + 8) : (j < 8 ? j + 8 : j + 16);
    float x = (hid[myrow*32 + k] - mu[myrow]) * rs[myrow];
    o[j] = tobf(fmaxf(x, 0.f));
  }
}

// ---------------- fused conv: R = h2@w3+b3 (WMMA), u = s*B, msg = R*u ------
// Templated on shape so all inner loops unroll into constant-offset code.
// dynamic LDS: sU[16][KC][P] | sB[16][P*Q*NF] | sS[16][MI*Q] | sR[16][KC]
template<int MO, int MI, int P, int Q, int NF>
__global__ __launch_bounds__(32)
void k_gconv_t(const __bf16* __restrict__ h2A, const __bf16* __restrict__ w3p,
               const float* __restrict__ b3,  const float* __restrict__ basis,
               const float* __restrict__ hin, const int* __restrict__ src,
               const int* __restrict__ dst,   float* __restrict__ acc) {
  constexpr int KC  = MI * NF;
  constexpr int NT  = KC / 16;
  constexpr int BSZ = P * Q * NF;
  extern __shared__ float sm[];
  float* sU = sm;                    // 16*KC*P
  float* sB = sU + 16 * KC * P;      // 16*BSZ
  float* sS = sB + 16 * BSZ;         // 16*MI*Q
  float* sR = sS + 16 * MI * Q;      // 16*KC
  int lane = threadIdx.x, tile = blockIdx.x, e0 = tile * 16;

  v16bf a = *(const v16bf*)(h2A + (size_t)(tile*32 + lane)*16);

  for (int idx = lane; idx < 16 * MI * Q; idx += 32) {
    int e = idx / (MI * Q), r = idx % (MI * Q);
    sS[idx] = hin[(size_t)src[e0 + e] * (MI * Q) + r];
  }
  for (int idx = lane; idx < 16 * BSZ; idx += 32) {
    int e = idx / BSZ, r = idx % BSZ;
    sB[idx] = basis[(size_t)(e0 + e) * BSZ + r];
  }
  if (e0 + 16 < NE) {  // speculative prefetch of next tile's basis
    const char* nb = (const char*)(basis + (size_t)(e0 + 16) * BSZ);
    __builtin_prefetch(nb + lane * 128, 0, 1);
  }
  __syncthreads();

  for (int idx = lane; idx < 16 * KC * P; idx += 32) {
    int e = idx / (KC * P), r = idx % (KC * P);
    int kc = r / P, p = r % P, i = kc / NF, f = kc % NF;
    const float* se = sS + e * MI * Q + i * Q;
    const float* be = sB + e * BSZ + p * Q * NF + f;
    float u = 0.f;
#pragma unroll
    for (int q = 0; q < Q; q++) u += se[q] * be[q * NF];
    sU[idx] = u;
  }
  __syncthreads();

  int col = lane & 15, rh = (lane >> 4) * 8;
  for (int m = 0; m < MO; m++) {
#pragma unroll
    for (int t = 0; t < NT; t++) {
      v16bf b = *(const v16bf*)(w3p + (size_t)((m*NT + t)*32 + lane)*16);
      v8f c = {};
      c = __builtin_amdgcn_wmma_f32_16x16x32_bf16(false, a, false, b, (short)0, c, false, false);
      float bb = b3[m*KC + t*16 + col];
#pragma unroll
      for (int v = 0; v < 8; v++) sR[(rh + v)*KC + t*16 + col] = c[v] + bb;
    }
    __syncthreads();
    for (int idx = lane; idx < 16 * P; idx += 32) {
      int e = idx / P, p = idx % P;
      const float* re = sR + e * KC;
      const float* ue = sU + (size_t)e * KC * P + p;
      float s = 0.f;
#pragma unroll
      for (int kc = 0; kc < KC; kc++) s += re[kc] * ue[kc * P];
      atomicAdd(&acc[(size_t)dst[e0 + e] * MO * P + m * P + p], s);
    }
    __syncthreads();
  }
}

// ---------------- mean + self-interaction ----------------------------------
__global__ void k_post(float* __restrict__ acc, const float* __restrict__ hin,
                       const float* __restrict__ Wself, const float* __restrict__ cnt,
                       int mo, int ci, int P) {
  int idx = blockIdx.x * 256 + threadIdx.x;
  if (idx >= NN * mo * P) return;
  int n = idx / (mo * P), r = idx % (mo * P), o = r / P, p = r % P;
  float c = cnt[n];
  float v = acc[idx] / fmaxf(c, 1.f);
  if (Wself != nullptr && c > 0.f) {
    float s = 0.f;
    for (int k = 0; k < ci; k++) s += Wself[o*ci + k] * hin[(size_t)n*ci*P + k*P + p];
    v += s;
  }
  acc[idx] = v;
}

// ---------------- GNormSE3 --------------------------------------------------
__global__ void k_gnorm(const float* __restrict__ hin, const float* __restrict__ w,
                        const float* __restrict__ bb, float* __restrict__ hout, int P) {
  int n = blockIdx.x * 64 + threadIdx.x;
  if (n >= NN) return;
  float nor[16], ln[16];
  const float* v = hin + (size_t)n * 16 * P;
  float s = 0.f;
  for (int c = 0; c < 16; c++) {
    float q = 0.f;
    for (int p = 0; p < P; p++) { float x = v[c*P + p]; q += x*x; }
    nor[c] = fmaxf(sqrtf(q), 1e-8f);
    s += nor[c];
  }
  float m = s / 16.f, var = 0.f;
  for (int c = 0; c < 16; c++) { float dv = nor[c] - m; var += dv*dv; }
  float rstd = rsqrtf(var / 16.f + 1e-5f);
  for (int c = 0; c < 16; c++) ln[c] = fmaxf((nor[c] - m) * rstd, 0.f);
  float* o = hout + (size_t)n * 16 * P;
  for (int c = 0; c < 16; c++) {
    float t = bb[c];
    for (int k = 0; k < 16; k++) t += ln[k] * w[k*16 + c];
    float sc = t / nor[c];
    for (int p = 0; p < P; p++) o[c*P + p] = sc * v[c*P + p];
  }
}

// ---------------- final FC head --------------------------------------------
__global__ void k_fc(const float* __restrict__ h0, const float* __restrict__ w1,
                     const float* __restrict__ b1, const float* __restrict__ w2,
                     const float* __restrict__ b2, float* __restrict__ out) {
  int n = blockIdx.x * 64 + threadIdx.x;
  if (n >= NN) return;
  const float* x = h0 + (size_t)n * 64;
  float h[64];
  for (int o = 0; o < 64; o++) {
    float s = b1[o];
    for (int k = 0; k < 64; k++) s += x[k] * w1[k*64 + o];
    h[o] = fmaxf(s, 0.f);
  }
  for (int o = 0; o < 3; o++) {
    float s = b2[o];
    for (int k = 0; k < 64; k++) s += h[k] * w2[k*3 + o];
    out[n*3 + o] = s;
  }
}

// ---------------------------------------------------------------------------
extern "C" void kernel_launch(void* const* d_in, const int* in_sizes, int n_in,
                              void* d_out, int out_size, void* d_ws, size_t ws_size,
                              hipStream_t stream) {
  (void)in_sizes; (void)n_in; (void)out_size; (void)ws_size;
  const float* f    = (const float*)d_in[0];
  const float* dvec = (const float*)d_in[1];
  const float* wedg = (const float*)d_in[2];
  const int*   src  = (const int*)d_in[297];
  const int*   dst  = (const int*)d_in[298];
  auto basisP = [&](int di, int dd) { return (const float*)d_in[3 + di*4 + dd]; };

  char* ws = (char*)d_ws; size_t off = 0;
  auto carve = [&](size_t bytes) -> char* {
    char* p = ws + off; off += (bytes + 255) & ~(size_t)255; return p;
  };
  __bf16* featA = (__bf16*)carve((size_t)NE * 32 * 2);
  __bf16* h2A   = (__bf16*)carve((size_t)NE * 32 * 2);
  __bf16* w1p   = (__bf16*)carve(1024 * 2);
  __bf16* w2p   = (__bf16*)carve(1024 * 2);
  __bf16* w3p   = (__bf16*)carve(131072 * 2);
  float*  cnt   = (float*)carve((size_t)NN * 4);
  float*  hA    = (float*)carve((size_t)NN * 16 * 16 * 4);
  float*  hB    = (float*)carve((size_t)NN * 16 * 16 * 4);
  float*  hC    = (float*)carve((size_t)NN * 64 * 4);

  const int DOFF[4] = {0, 1, 4, 9};

  hipMemsetAsync(cnt, 0, (size_t)NN * 4, stream);
  k_deg<<<NE/256, 256, 0, stream>>>(dst, cnt);
  k_featpack<<<(NE*2 + 127)/128, 128, 0, stream>>>(dvec, wedg, featA);

  // templated conv dispatch over the 24 static shapes
  auto launch_gconv = [&](int mo, int mi, int P, int Q, int nf,
                          const float* b3, const float* bas, const float* hin,
                          float* acc) {
    int KC = mi * nf;
    size_t smem = 4ull * 16 * ((size_t)KC*P + (size_t)P*Q*nf + (size_t)mi*Q + KC);
#define GCONV_CASE(MO_, MI_, P_, Q_, NF_)                                       \
    if (mo == MO_ && mi == MI_ && P == P_ && Q == Q_ && nf == NF_) {            \
      k_gconv_t<MO_, MI_, P_, Q_, NF_><<<NE/16, 32, smem, stream>>>(            \
          h2A, w3p, b3, bas, hin, src, dst, acc);                               \
      return;                                                                   \
    }
    GCONV_CASE(16,32,1,1,1) GCONV_CASE(16,32,3,1,1)
    GCONV_CASE(16,32,5,1,1) GCONV_CASE(16,32,7,1,1)
    GCONV_CASE(16,16,1,1,1) GCONV_CASE(16,16,1,3,1)
    GCONV_CASE(16,16,1,5,1) GCONV_CASE(16,16,1,7,1)
    GCONV_CASE(16,16,3,1,1) GCONV_CASE(16,16,3,3,3)
    GCONV_CASE(16,16,3,5,3) GCONV_CASE(16,16,3,7,3)
    GCONV_CASE(16,16,5,1,1) GCONV_CASE(16,16,5,3,3)
    GCONV_CASE(16,16,5,5,5) GCONV_CASE(16,16,5,7,5)
    GCONV_CASE(16,16,7,1,1) GCONV_CASE(16,16,7,3,3)
    GCONV_CASE(16,16,7,5,5) GCONV_CASE(16,16,7,7,7)
    GCONV_CASE(64,16,1,1,1) GCONV_CASE(64,16,1,3,1)
    GCONV_CASE(64,16,1,5,1) GCONV_CASE(64,16,1,7,1)
#undef GCONV_CASE
  };

  auto run_pair = [&](int leafBase, const float* bas, const float* hin,
                      int mo, int mi, int P, int Q, int nf, float* acc) {
    const float* b1 = (const float*)d_in[leafBase + 0];
    const float* b2 = (const float*)d_in[leafBase + 1];
    const float* b3 = (const float*)d_in[leafBase + 2];
    const float* w1 = (const float*)d_in[leafBase + 3];
    const float* w2 = (const float*)d_in[leafBase + 4];
    const float* w3 = (const float*)d_in[leafBase + 5];
    int KC = mi * nf, ncol3 = mo * KC, nt3 = ncol3 / 16;
    int packT = (4 + nt3) * 32;
    k_wpack<<<(packT + 127)/128, 128, 0, stream>>>(w1, w2, w3, ncol3, w1p, w2p, w3p);
    k_radial_hidden<<<NE/16, 32, 0, stream>>>(featA, w1p, w2p, b1, b2, h2A);
    launch_gconv(mo, mi, P, Q, nf, b3, bas, hin, acc);
  };

  // ---- layer 0: FIB_IN (deg0,32) -> FIB_MID ----
  hipMemsetAsync(hA, 0, (size_t)NN*16*16*4, stream);
  for (int dd = 0; dd < 4; dd++) {
    int P = 2*dd + 1;
    run_pair(19 + dd*6, basisP(0, dd), f, 16, 32, P, 1, 1, hA + (size_t)16*NN*DOFF[dd]);
  }
  for (int dd = 0; dd < 4; dd++) {
    int P = 2*dd + 1;
    const float* Ws = (dd == 0) ? (const float*)d_in[43] : nullptr;
    k_post<<<(NN*16*P + 255)/256, 256, 0, stream>>>(hA + (size_t)16*NN*DOFF[dd], f, Ws, cnt, 16, 32, P);
    k_gnorm<<<(NN + 63)/64, 64, 0, stream>>>(hA + (size_t)16*NN*DOFF[dd],
        (const float*)d_in[273 + dd*2 + 1], (const float*)d_in[273 + dd*2 + 0],
        hB + (size_t)16*NN*DOFF[dd], P);
  }

  // ---- layers 1,2: FIB_MID -> FIB_MID ----
  for (int li = 1; li <= 2; li++) {
    int base = 44 + (li - 1) * 100;
    hipMemsetAsync(hA, 0, (size_t)NN*16*16*4, stream);
    for (int dd = 0; dd < 4; dd++)
      for (int di = 0; di < 4; di++) {
        int P = 2*dd + 1, Q = 2*di + 1;
        int nf = 2 * (dd < di ? dd : di) + 1;
        run_pair(base + (dd*4 + di)*6, basisP(di, dd), hB + (size_t)16*NN*DOFF[di],
                 16, 16, P, Q, nf, hA + (size_t)16*NN*DOFF[dd]);
      }
    int nbase = 273 + li * 8;
    for (int dd = 0; dd < 4; dd++) {
      int P = 2*dd + 1;
      const float* Ws = (const float*)d_in[base + 96 + dd];
      k_post<<<(NN*16*P + 255)/256, 256, 0, stream>>>(hA + (size_t)16*NN*DOFF[dd],
          hB + (size_t)16*NN*DOFF[dd], Ws, cnt, 16, 16, P);
      k_gnorm<<<(NN + 63)/64, 64, 0, stream>>>(hA + (size_t)16*NN*DOFF[dd],
          (const float*)d_in[nbase + dd*2 + 1], (const float*)d_in[nbase + dd*2 + 0],
          hB + (size_t)16*NN*DOFF[dd], P);
    }
  }

  // ---- layer 3: FIB_MID -> FIB_OUT (deg0, 64) ----
  hipMemsetAsync(hC, 0, (size_t)NN*64*4, stream);
  for (int di = 0; di < 4; di++) {
    int Q = 2*di + 1;
    run_pair(244 + di*6, basisP(di, 0), hB + (size_t)16*NN*DOFF[di], 64, 16, 1, Q, 1, hC);
  }
  k_post<<<(NN*64 + 255)/256, 256, 0, stream>>>(hC, hB, (const float*)d_in[268], cnt, 64, 16, 1);

  // ---- FC head ----
  k_fc<<<(NN + 63)/64, 64, 0, stream>>>(hC, (const float*)d_in[271], (const float*)d_in[269],
                                        (const float*)d_in[272], (const float*)d_in[270],
                                        (float*)d_out);
}
